// CrossWindowAttention_24094766530835
// MI455X (gfx1250) — compile-verified, matrix operand order checked by reading, not run on hardware
//
#include <hip/hip_runtime.h>

// CDNA5 / gfx1250 windowed cross-attention (Swin-style), wave32 + WMMA bf16.
// b=8192 windows, n=49 tokens (pad->64), c=96, heads=3, hd=32.

typedef __bf16 bf16_t;
typedef __attribute__((ext_vector_type(4)))  __bf16 v4bf;
typedef __attribute__((ext_vector_type(8)))  __bf16 v8bf;
typedef __attribute__((ext_vector_type(16))) __bf16 v16bf;
typedef __attribute__((ext_vector_type(8)))  float  v8f;

#define WIN_N 49
#define DIMC  96
#define HEADS 3
#define HD    32
#define QSCALE 0.17677669529663687f   // 1/sqrt(32), folded into Wq/bq at prep

__device__ __forceinline__ bf16_t f2bf(float f) { return (bf16_t)f; }

__device__ __forceinline__ v8f vzero() {
  v8f z = {0.f, 0.f, 0.f, 0.f, 0.f, 0.f, 0.f, 0.f};
  return z;
}

// Load one 16x32 bf16 A- or B-fragment from row-major storage (stride in
// elements). Lane l (l<16): row = row0+l, K chunks [k0+koff, +8) and
// [k0+koff+16, +8) with koff = (l>=16)*8 — CDNA5 16-bit WMMA layout.
// Each chunk is one contiguous 16-byte load (ds_load_b128 / global_load_b128).
__device__ __forceinline__ v16bf load_frag(const bf16_t* base, int row0, int k0, int stride) {
  const int l = threadIdx.x & 31;
  const int r = row0 + (l & 15);
  const int k = k0 + ((l >> 4) << 3);
  const bf16_t* p = base + r * stride + k;
  union { v16bf v; v8bf h[2]; } u;
  u.h[0] = *(const v8bf*)(p);
  u.h[1] = *(const v8bf*)(p + 16);
  return u.v;
}

__device__ __forceinline__ v8f wmma_bf16(v16bf a, v16bf b, v8f c) {
  return __builtin_amdgcn_wmma_f32_16x16x32_bf16(false, a, false, b, (short)0, c, false, false);
}

// ---------------------------------------------------------------------------
// Prep kernel: weights -> bf16 transposed (B-fragment friendly: T[n][k]),
// packed contiguously as W3T[3][96][96] (Q pre-scaled by 1/sqrt(hd)), plus
// WpT[96][96], padded bias table [3][64][64] (-1e30 key mask), and the three
// projection bias vectors packed as bvec[3][96] (Q slice pre-scaled).
// ---------------------------------------------------------------------------
__global__ __launch_bounds__(256) void xwin_prep(
    const float* __restrict__ w_q, const float* __restrict__ w_kv,
    const float* __restrict__ w_proj,
    const float* __restrict__ b_q, const float* __restrict__ b_kv,
    const float* __restrict__ rpb, const int* __restrict__ rel,
    bf16_t* __restrict__ W3T, bf16_t* __restrict__ WpT,
    float* __restrict__ biasT, float* __restrict__ bvec) {
  int gid = blockIdx.x * 256 + threadIdx.x;   // 194 blocks cover 49440 tasks
  if (gid < 27648) {                          // W3T: mat-major [3][96][96]
    int mat = gid / 9216, i = gid % 9216;
    int n = i / 96, k = i % 96;
    float w;
    if (mat == 0)      w = w_q[k * 96 + n] * QSCALE;
    else if (mat == 1) w = w_kv[k * 192 + n];
    else               w = w_kv[k * 192 + 96 + n];
    W3T[gid] = f2bf(w);
  } else if (gid < 36864) {                   // WpT
    int i = gid - 27648, n = i / 96, k = i % 96;
    WpT[i] = f2bf(w_proj[k * 96 + n]);
  } else if (gid < 49152) {                   // padded attn bias [3][64][64]
    int i = gid - 36864;
    int h = i / 4096, r = (i / 64) & 63, j = i & 63;
    float bv;
    if (j >= WIN_N)      bv = -1e30f;         // mask padded keys
    else if (r >= WIN_N) bv = 0.0f;           // padded queries: dont-care
    else                 bv = rpb[rel[r * WIN_N + j] * HEADS + h];
    biasT[i] = bv;
  } else if (gid < 49440) {                   // bvec[3][96]
    int i = gid - 49152, mat = i / 96, n = i % 96;
    float bv;
    if (mat == 0)      bv = b_q[n] * QSCALE;
    else if (mat == 1) bv = b_kv[n];
    else               bv = b_kv[96 + n];
    bvec[i] = bv;
  }
}

// ---------------------------------------------------------------------------
// Main kernel: one window per block, 8 wave32s.
// ---------------------------------------------------------------------------
__global__ __launch_bounds__(256) void xwin_attn(
    const float* __restrict__ x, const float* __restrict__ v,
    const float* __restrict__ b_proj,
    const bf16_t* __restrict__ W3T, const bf16_t* __restrict__ WpT,
    const float* __restrict__ biasT, const float* __restrict__ bvec,
    float* __restrict__ out) {
  // LDS plan (60 KB, hand-aliased):
  //  [0,     12288) Xb  [64][96] bf16   } stage 1-2, later aliased by:
  //  [12288, 24576) Vb  [64][96] bf16   }   S [64][64] f32 @0 (P bf16 in-place)
  //                                         Oh [64][32] bf16 @16384
  //  [24576, 36864) Qb  [64][96] bf16
  //  [36864, 49152) Kb  [64][96] bf16
  //  [49152, 61440) VvT [96][64] bf16  (transposed V-projection)
  __shared__ __align__(16) unsigned char smem[61440];
  bf16_t* Xb  = (bf16_t*)(smem);
  bf16_t* Vb  = (bf16_t*)(smem + 12288);
  float*  S   = (float*)(smem);              // aliases Xb (after stage 2)
  bf16_t* Oh  = (bf16_t*)(smem + 16384);     // aliases Vb tail
  bf16_t* Qb  = (bf16_t*)(smem + 24576);
  bf16_t* Kb  = (bf16_t*)(smem + 36864);
  bf16_t* VvT = (bf16_t*)(smem + 49152);

  const int tid  = threadIdx.x;
  const int wave = tid >> 5;
  const int lane = tid & 31;
  const size_t base = (size_t)blockIdx.x * (WIN_N * DIMC);

  // ---- Stage 1: stage x,v into LDS as bf16, zero-pad rows 49..63 ----------
#pragma unroll
  for (int it = 0; it < 6; ++it) {            // 6*256 = 1536 float4 chunks
    int c4 = tid + it * 256;
    int row = c4 / 24, col = (c4 % 24) * 4;
    float4 xv = {0.f, 0.f, 0.f, 0.f};
    float4 vv = {0.f, 0.f, 0.f, 0.f};
    if (row < WIN_N) {
      xv = *(const float4*)(x + base + row * DIMC + col);
      vv = *(const float4*)(v + base + row * DIMC + col);
    }
    int o = row * DIMC + col;
    v4bf xp = {f2bf(xv.x), f2bf(xv.y), f2bf(xv.z), f2bf(xv.w)};
    v4bf vp = {f2bf(vv.x), f2bf(vv.y), f2bf(vv.z), f2bf(vv.w)};
    *(v4bf*)(Xb + o) = xp;                    // one ds_store_b64 each
    *(v4bf*)(Vb + o) = vp;
  }
  __syncthreads();

  // ---- Stage 2: [Q|K|Vv] = [X|V|V] * W3T (+bvec); 72 tiles, K=96 ----------
  // Constant trip count (9) -> fully unrolled, 27 static WMMAs.
#pragma unroll
  for (int it = 0; it < 9; ++it) {
    int task = wave + it * 8;
    int mat = task / 24;                 // 0=Q, 1=K, 2=Vv
    int rem = task % 24;
    int mt = rem & 3, nt = rem >> 2;
    const bf16_t* A = (mat == 0) ? Xb : Vb;
    const bf16_t* B = W3T + mat * 9216;  // contiguous: one MAD, no selects
    v8f acc = vzero();
#pragma unroll
    for (int kb = 0; kb < 96; kb += 32) {
      acc = wmma_bf16(load_frag(A, mt * 16, kb, DIMC),
                      load_frag(B, nt * 16, kb, DIMC), acc);
    }
    int n  = nt * 16 + (lane & 15);
    int m0 = mt * 16 + ((lane >> 4) << 3);
    float bias_v = bvec[mat * 96 + n];   // single indexed load, no selects
    if (mat == 2) {                      // store transposed: VvT[n][m0..m0+7]
      v8bf pk;
#pragma unroll
      for (int i = 0; i < 8; i++) pk[i] = f2bf(acc[i] + bias_v);
      *(v8bf*)(VvT + n * 64 + m0) = pk;  // one ds_store_b128
    } else {                             // Q and K epilogues now identical
      bf16_t* Dst = (mat == 0) ? Qb : Kb;
#pragma unroll
      for (int i = 0; i < 8; i++) {
        Dst[(m0 + i) * DIMC + n] = f2bf(acc[i] + bias_v);
      }
    }
  }
  __syncthreads();

  // Persistent out-projection accumulators: 3 Y-tiles per wave (4x6 grid).
  v8f yacc[3];
#pragma unroll
  for (int j = 0; j < 3; j++) yacc[j] = vzero();

  for (int h = 0; h < HEADS; ++h) {
    // ---- Stage 3: S = Q_h * K_h^T + bias; 16 tiles, K=32 (one WMMA) -------
#pragma unroll
    for (int it = 0; it < 2; ++it) {
      int t = wave + it * 8;
      int mt = t & 3, nt = t >> 2;
      v8f acc = wmma_bf16(load_frag(Qb, mt * 16, h * HD, DIMC),
                          load_frag(Kb, nt * 16, h * HD, DIMC), vzero());
      int n  = nt * 16 + (lane & 15);
      int m0 = mt * 16 + ((lane >> 4) << 3);
#pragma unroll
      for (int i = 0; i < 8; i++) {
        S[(m0 + i) * 64 + n] = acc[i] + biasT[(h * 64 + m0 + i) * 64 + n];
      }
    }
    __syncthreads();

    // ---- Stage 4: row softmax, P written bf16 in place over S -------------
    if (tid < 64) {
      float* srow = S + tid * 64;
      bf16_t* prow = (bf16_t*)srow;       // in-place, ascending j is safe
      if (tid < WIN_N) {
        float mx = -3.4e38f;
        for (int j = 0; j < 64; j++) mx = fmaxf(mx, srow[j]);
        float sum = 0.f;
        for (int j = 0; j < 64; j++) {
          float e = __expf(srow[j] - mx);
          sum += e;
          prow[j] = f2bf(e);
        }
        float inv = 1.f / sum;
        for (int j = 0; j < 64; j++) prow[j] = f2bf((float)prow[j] * inv);
      } else {
        for (int j = 0; j < 64; j++) prow[j] = f2bf(0.f);
      }
    }
    __syncthreads();

    // ---- Stage 5: O_h = P * Vv_h; 8 tiles (4x2), one per wave, K=64 -------
    {
      int mt = wave & 3, nt = wave >> 2;
      v8f acc = vzero();
#pragma unroll
      for (int kb = 0; kb < 64; kb += 32) {
        acc = wmma_bf16(load_frag((const bf16_t*)S, mt * 16, kb, 128),
                        load_frag(VvT, h * HD + nt * 16, kb, 64), acc);
      }
      int n  = nt * 16 + (lane & 15);
      int m0 = mt * 16 + ((lane >> 4) << 3);
#pragma unroll
      for (int i = 0; i < 8; i++) Oh[(m0 + i) * HD + n] = f2bf(acc[i]);
    }
    __syncthreads();

    // ---- Stage 6: Y += O_h * Wp[h*32:(h+1)*32, :]; K=32 per head ----------
#pragma unroll
    for (int j = 0; j < 3; j++) {
      int id = wave * 3 + j;
      int mt = id & 3, nt = id >> 2;
      yacc[j] = wmma_bf16(load_frag(Oh, mt * 16, 0, HD),
                          load_frag(WpT, nt * 16, h * HD, DIMC), yacc[j]);
    }
    __syncthreads();
  }

  // ---- Final: Y + b_proj -> global (rows < 49 only) -----------------------
#pragma unroll
  for (int j = 0; j < 3; j++) {
    int id = wave * 3 + j;
    int mt = id & 3, nt = id >> 2;
    int n  = nt * 16 + (lane & 15);
    int m0 = mt * 16 + ((lane >> 4) << 3);
    float bp = b_proj[n];
#pragma unroll
    for (int i = 0; i < 8; i++) {
      int m = m0 + i;
      if (m < WIN_N) out[base + m * DIMC + n] = yacc[j][i] + bp;
    }
  }
}

// ---------------------------------------------------------------------------
extern "C" void kernel_launch(void* const* d_in, const int* in_sizes, int n_in,
                              void* d_out, int out_size, void* d_ws, size_t ws_size,
                              hipStream_t stream) {
  const float* x      = (const float*)d_in[0];
  const float* v      = (const float*)d_in[1];
  const float* w_q    = (const float*)d_in[2];
  const float* b_q    = (const float*)d_in[3];
  const float* w_kv   = (const float*)d_in[4];
  const float* b_kv   = (const float*)d_in[5];
  const float* w_proj = (const float*)d_in[6];
  const float* b_proj = (const float*)d_in[7];
  const float* rpb    = (const float*)d_in[8];
  const int*   rel    = (const int*)d_in[9];

  // Workspace layout (124,032 bytes):
  //  [0,      55296) W3T  [3][96][96] bf16 (Q slice pre-scaled)
  //  [55296,  73728) WpT  [96][96]    bf16
  //  [73728, 122880) bias [3][64][64] f32
  //  [122880,124032) bvec [3][96]     f32
  char* ws = (char*)d_ws;
  bf16_t* W3T  = (bf16_t*)(ws + 0);
  bf16_t* WpT  = (bf16_t*)(ws + 55296);
  float*  bias = (float*)(ws + 73728);
  float*  bvec = (float*)(ws + 122880);

  xwin_prep<<<194, 256, 0, stream>>>(w_q, w_kv, w_proj, b_q, b_kv, rpb, rel,
                                     W3T, WpT, bias, bvec);
  xwin_attn<<<8192, 256, 0, stream>>>(x, v, b_proj, W3T, WpT, bias, bvec,
                                      (float*)d_out);
}
